// LSTM_59966333387334
// MI455X (gfx1250) — matrix operand order repaired and checked by reading, not compiled, not run on hardware
//
#include <hip/hip_runtime.h>
#include <hip/hip_bf16.h>
#include <cstdint>

#define DEVI __device__ __forceinline__

typedef __attribute__((ext_vector_type(16))) __bf16 v16bf;
typedef __attribute__((ext_vector_type(8)))  float  v8f;
typedef __attribute__((ext_vector_type(4)))  int    v4i;

static constexpr int B   = 64;
static constexpr int S   = 512;
static constexpr int I   = 64;
static constexpr int H   = 1024;
static constexpr int G4H = 4 * H;      // 4096 gate rows
static constexpr int KTH = H / 32;     // 32 k-steps for the H-sized GEMM

DEVI unsigned short f2bf(float f) {
  unsigned int u = __float_as_uint(f);
  u += 0x7fffu + ((u >> 16) & 1u);     // round to nearest even
  return (unsigned short)(u >> 16);
}
DEVI float sigf(float x)   { return 1.0f / (1.0f + __expf(-x)); }
DEVI float tanhf_f(float x){ return 1.0f - 2.0f / (__expf(2.0f * x) + 1.0f); }

// ---- async global->LDS staging (CDNA5), with sync fallback ----
#if defined(__has_builtin)
#if __has_builtin(__builtin_amdgcn_global_load_async_to_lds_b128)
#define HAVE_ASYNC_LDS 1
#endif
#endif

#ifdef HAVE_ASYNC_LDS
typedef __attribute__((address_space(1))) v4i* g_v4i_p;
typedef __attribute__((address_space(3))) v4i* l_v4i_p;
#endif

DEVI void stage32(const unsigned short* g, unsigned short* l) {
#ifdef HAVE_ASYNC_LDS
  // two B128 async copies: 32 bytes per lane, no VGPR round trip
  __builtin_amdgcn_global_load_async_to_lds_b128(
      (g_v4i_p)(g), (l_v4i_p)(l), 0, 0);
  __builtin_amdgcn_global_load_async_to_lds_b128(
      (g_v4i_p)(g + 8), (l_v4i_p)(l + 8), 0, 0);
#else
  *(uint4*)l       = *(const uint4*)g;
  *(uint4*)(l + 8) = *(const uint4*)(g + 8);
#endif
}

DEVI void wait_async() {
#ifdef HAVE_ASYNC_LDS
#if __has_builtin(__builtin_amdgcn_s_wait_asynccnt)
  __builtin_amdgcn_s_wait_asynccnt(0);
#else
  asm volatile("s_wait_asynccnt 0x0" ::: "memory");
#endif
#endif
}

// ---------------- helper kernels ----------------

__global__ void zero_ws_kernel(unsigned int* p, int n) {
  int i = blockIdx.x * blockDim.x + threadIdx.x;
  if (i < n) p[i] = 0u;
}

__global__ void cvt_f32_bf16_kernel(const float* __restrict__ src,
                                    unsigned short* __restrict__ dst, int n) {
  int i = blockIdx.x * blockDim.x + threadIdx.x;
  if (i < n) dst[i] = f2bf(src[i]);
}

// Pack W[G4H x K] (fp32, row-major) into per-tile WMMA-B layout:
// tile (nt, kt) covers rows nt*16..+16, cols kt*32..+32.
// Lane L owns 16 contiguous bf16: row nt*16+(L&15),
// cols kt*32 + (L>=16 ? 16 : 0) .. +16   (same contiguous k order as the A fragment).
__global__ void pack_w_kernel(const float* __restrict__ W,
                              unsigned short* __restrict__ P,
                              int ktiles, int K) {
  int t    = blockIdx.x * blockDim.x + threadIdx.x;
  int lane = t & 31;
  int tile = t >> 5;
  int total = (G4H / 16) * ktiles;
  if (tile >= total) return;
  int nt = tile / ktiles;
  int kt = tile % ktiles;
  int row   = nt * 16 + (lane & 15);
  int kbase = kt * 32 + ((lane >> 4) << 4);
  const float* src = W + (size_t)row * K + kbase;
  unsigned short* dst = P + (size_t)tile * 512 + lane * 16;
#pragma unroll
  for (int e = 0; e < 16; ++e) dst[e] = f2bf(src[e]);
}

// ---------------- fused LSTM step (one layer, one timestep) ----------------
// Block = 8 waves = {4 M-tiles} x {2 N-tiles}. Per k-step the block stages the
// 8 required B tiles (2 nt x 4 gates, 8KB) into LDS (double-buffered) with
// async global->LDS copies; each B tile is then read by the 4 waves that share
// its nt, cutting L2 weight traffic 4x. A (activations) streams from global.
__global__ __launch_bounds__(256, 4) void lstm_step_kernel(
    const unsigned short* __restrict__ Ax, int pitchX, int ktX,
    const unsigned short* __restrict__ Wxp,
    const unsigned short* __restrict__ Ah,
    const unsigned short* __restrict__ Whp,
    const float* __restrict__ bias,
    float* __restrict__ c,
    float* __restrict__ hf,
    unsigned short* __restrict__ hb) {
  __shared__ unsigned short smem[2 * 8 * 512];   // 2 buffers x 8 tiles x 1KB

  const int tid  = threadIdx.x;
  const int lane = tid & 31;
  const int wv   = tid >> 5;                 // 0..7
  const int mt   = wv & 3;                   // M tile
  const int ntl  = wv >> 2;                  // local N tile 0..1
  const int ntg  = (blockIdx.x << 1) + ntl;  // global N tile 0..63
  const int lr   = lane & 15;
  const int hi   = lane >> 4;

  // staging role: wave wv copies tile slot wv = (sntl*4 + sg), 32B per lane
  const int sg   = wv & 3;                   // gate
  const int sntl = wv >> 2;                  // local nt
  const int sntg = (blockIdx.x << 1) + sntl;

  v8f acc0 = {}, acc1 = {}, acc2 = {}, acc3 = {};

  for (int phase = 0; phase < 2; ++phase) {
    const unsigned short* A  = phase ? Ah  : Ax;
    const int pitch          = phase ? H   : pitchX;
    const int kt             = phase ? KTH : ktX;
    const unsigned short* Wp = phase ? Whp : Wxp;

    const unsigned short* arow =
        A + (size_t)((mt << 4) + lr) * pitch + (hi << 4);
    // staging source: tiles of gate sg / column tile sntg, consecutive in ks
    const unsigned short* sbase =
        Wp + (size_t)(sg * 64 + sntg) * kt * 512 + lane * 16;
    unsigned short* sdst = &smem[wv * 512 + lane * 16];  // slot in buffer 0

    __syncthreads();              // protect previous phase's LDS reads
    stage32(sbase, sdst);         // ks = 0 into buffer 0

    for (int ks = 0; ks < kt; ++ks) {
      const int cur = ks & 1;
      wait_async();
      __syncthreads();            // buf[cur] ready; prior reads of buf[cur^1] done
      if (ks + 1 < kt) {
        __builtin_prefetch(sbase + (size_t)(ks + 3) * 512, 0, 1);
        stage32(sbase + (size_t)(ks + 1) * 512, sdst + (cur ^ 1) * 4096);
      }

      // A fragment (16x32 bf16): two 16B global loads per lane
      union { v16bf bf; uint4 q[2]; } fa;
      const unsigned short* ap = arow + ks * 32;
      fa.q[0] = *(const uint4*)(ap);
      fa.q[1] = *(const uint4*)(ap + 8);

      // B fragments from LDS: 4 gate tiles for this wave's nt
      const unsigned short* bl = &smem[cur * 4096 + (ntl << 2) * 512 + lane * 16];
      union { v16bf bf; uint4 q[2]; } fb;
      fb.q[0] = *(const uint4*)(bl);
      fb.q[1] = *(const uint4*)(bl + 8);
      acc0 = __builtin_amdgcn_wmma_f32_16x16x32_bf16(false, fa.bf, false, fb.bf,
                                                     (short)0, acc0, false, false);
      fb.q[0] = *(const uint4*)(bl + 512);
      fb.q[1] = *(const uint4*)(bl + 512 + 8);
      acc1 = __builtin_amdgcn_wmma_f32_16x16x32_bf16(false, fa.bf, false, fb.bf,
                                                     (short)0, acc1, false, false);
      fb.q[0] = *(const uint4*)(bl + 1024);
      fb.q[1] = *(const uint4*)(bl + 1024 + 8);
      acc2 = __builtin_amdgcn_wmma_f32_16x16x32_bf16(false, fa.bf, false, fb.bf,
                                                     (short)0, acc2, false, false);
      fb.q[0] = *(const uint4*)(bl + 1536);
      fb.q[1] = *(const uint4*)(bl + 1536 + 8);
      acc3 = __builtin_amdgcn_wmma_f32_16x16x32_bf16(false, fa.bf, false, fb.bf,
                                                     (short)0, acc3, false, false);
    }
  }

  // Bias + activations + cell update. C/D layout: VGPR v -> row mt*16+hi*8+v,
  // col ntg*16+lr.
  const int col = (ntg << 4) + lr;
  const float b0v = bias[col];
  const float b1v = bias[H + col];
  const float b2v = bias[2 * H + col];
  const float b3v = bias[3 * H + col];
  const int rbase = (mt << 4) + (hi << 3);
#pragma unroll
  for (int v = 0; v < 8; ++v) {
    float ig = sigf(acc0[v] + b0v);
    float fg = sigf(acc1[v] + b1v);
    float gg = tanhf_f(acc2[v] + b2v);
    float og = sigf(acc3[v] + b3v);
    int idx = (rbase + v) * H + col;
    float cn = fg * c[idx] + ig * gg;
    c[idx] = cn;
    float hv = og * tanhf_f(cn);
    hf[idx] = hv;
    hb[idx] = f2bf(hv);
  }
}

// ---------------- tiny FC head ----------------
__global__ void fc_kernel(const float* __restrict__ hfin,
                          const float* __restrict__ W,
                          const float* __restrict__ bias,
                          float* __restrict__ out) {
  int t = blockIdx.x * blockDim.x + threadIdx.x;
  if (t >= B * 12) return;
  int b = t / 12, j = t % 12;
  const float* hr = hfin + (size_t)b * H;
  const float* wr = W + (size_t)j * H;
  float s = 0.f;
  for (int k = 0; k < H; k += 4)
    s += hr[k] * wr[k] + hr[k + 1] * wr[k + 1] + hr[k + 2] * wr[k + 2] +
         hr[k + 3] * wr[k + 3];
  out[t] = s + bias[j];
}

// ---------------- host launch ----------------
extern "C" void kernel_launch(void* const* d_in, const int* in_sizes, int n_in,
                              void* d_out, int out_size, void* d_ws, size_t ws_size,
                              hipStream_t stream) {
  const float* x   = (const float*)d_in[0];
  const float* Wi0 = (const float*)d_in[1];
  const float* Wh0 = (const float*)d_in[2];
  const float* b0  = (const float*)d_in[3];
  const float* Wi1 = (const float*)d_in[4];
  const float* Wh1 = (const float*)d_in[5];
  const float* b1  = (const float*)d_in[6];
  const float* fcW = (const float*)d_in[7];
  const float* fcb = (const float*)d_in[8];

  char* ws = (char*)d_ws;
  size_t off = 0;
  auto take = [&](size_t bytes) -> char* {
    char* p = ws + off;
    off = (off + bytes + 255) & ~(size_t)255;
    return p;
  };

  unsigned short* xbf  = (unsigned short*)take((size_t)B * S * I * 2);
  unsigned short* Wi0p = (unsigned short*)take((size_t)G4H * I * 2);
  unsigned short* Wh0p = (unsigned short*)take((size_t)G4H * H * 2);
  unsigned short* Wi1p = (unsigned short*)take((size_t)G4H * H * 2);
  unsigned short* Wh1p = (unsigned short*)take((size_t)G4H * H * 2);

  char* stateBegin = ws + off;
  float* c0  = (float*)take((size_t)B * H * 4);
  float* c1  = (float*)take((size_t)B * H * 4);
  float* h0f = (float*)take((size_t)B * H * 4);
  float* h1f = (float*)take((size_t)B * H * 4);
  unsigned short* h0b[2] = {(unsigned short*)take((size_t)B * H * 2),
                            (unsigned short*)take((size_t)B * H * 2)};
  unsigned short* h1b[2] = {(unsigned short*)take((size_t)B * H * 2),
                            (unsigned short*)take((size_t)B * H * 2)};
  size_t stateBytes = (size_t)((ws + off) - stateBegin);

  // 1) zero recurrent state (h, c) every call — deterministic.
  {
    int n = (int)(stateBytes / 4);
    zero_ws_kernel<<<(n + 255) / 256, 256, 0, stream>>>((unsigned int*)stateBegin, n);
  }
  // 2) x -> bf16
  {
    int n = B * S * I;
    cvt_f32_bf16_kernel<<<(n + 255) / 256, 256, 0, stream>>>(x, xbf, n);
  }
  // 3) pack weights into WMMA-B tile layout (bf16)
  pack_w_kernel<<<((G4H / 16) * 2 * 32 + 255) / 256, 256, 0, stream>>>(Wi0, Wi0p, 2, I);
  pack_w_kernel<<<((G4H / 16) * KTH * 32 + 255) / 256, 256, 0, stream>>>(Wh0, Wh0p, KTH, H);
  pack_w_kernel<<<((G4H / 16) * KTH * 32 + 255) / 256, 256, 0, stream>>>(Wi1, Wi1p, KTH, H);
  pack_w_kernel<<<((G4H / 16) * KTH * 32 + 255) / 256, 256, 0, stream>>>(Wh1, Wh1p, KTH, H);

  // 4) sequential scan: 2 launches per timestep (global sync via launch order),
  //    double-buffered h so h(t) readers never race h(t+1) writers.
  for (int t = 0; t < S; ++t) {
    int par = t & 1;
    // layer 0: Ax = x_t (row pitch S*I), Kx = 64 -> 2 k-steps
    lstm_step_kernel<<<32, 256, 0, stream>>>(
        xbf + (size_t)t * I, S * I, 2, Wi0p,
        h0b[par], Wh0p, b0, c0, h0f, h0b[par ^ 1]);
    // layer 1: Ax = h0(t) just produced, Kx = 1024 -> 32 k-steps
    lstm_step_kernel<<<32, 256, 0, stream>>>(
        h0b[par ^ 1], H, KTH, Wi1p,
        h1b[par], Wh1p, b1, c1, h1f, h1b[par ^ 1]);
  }

  // 5) FC head on final h1 (fp32)
  fc_kernel<<<3, 256, 0, stream>>>(h1f, fcW, fcb, (float*)d_out);
}
